// Qwen3Attention_60481729463035
// MI455X (gfx1250) — matrix operand order repaired and checked by reading, not compiled, not run on hardware
//
#include <hip/hip_runtime.h>
#include <cstdint>
#include <cstddef>

// ---------------------------------------------------------------------------
// Qwen3 attention block for gfx1250 (MI455X), bf16 hi/lo split-precision WMMA
// with double-buffered LDS tiles staged via GLOBAL_LOAD_ASYNC_TO_LDS_B128.
// ---------------------------------------------------------------------------

#define T_    2048
#define HID   4096
#define NQH   32
#define NKVH  8
#define HD    128
#define QKVW  6144            // Q(4096) + K(1024) + V(1024)
#define SCALE 0.088388347648318447f   // 128^-0.5

typedef __bf16 bf16_t;
typedef __attribute__((ext_vector_type(16))) bf16_t          bf16x16;
typedef __attribute__((ext_vector_type(8)))  float           f32x8;
typedef __attribute__((ext_vector_type(8)))  unsigned short  u16x8;
typedef __attribute__((ext_vector_type(4)))  unsigned short  u16x4;
typedef int v4i_vs __attribute__((vector_size(16)));   // matches builtin param

#if __has_builtin(__builtin_amdgcn_global_load_async_to_lds_b128)
#define ASYNC_LDS 1
#else
#define ASYNC_LDS 0
#endif

// ---- helpers ---------------------------------------------------------------

__device__ __forceinline__ unsigned short f32_to_bf16_rne(float f) {
  union { float f; unsigned int u; } v; v.f = f;
  unsigned int u = v.u;
  unsigned int r = u + 0x7FFFu + ((u >> 16) & 1u);   // round-to-nearest-even
  return (unsigned short)(r >> 16);
}
__device__ __forceinline__ float bf16_bits_to_f32(unsigned short h) {
  union { unsigned int u; float f; } v; v.u = ((unsigned int)h) << 16;
  return v.f;
}
__device__ __forceinline__ bf16x16 frag16(u16x8 a, u16x8 b) {
  union { u16x8 v[2]; bf16x16 f; } u;
  u.v[0] = a; u.v[1] = b;
  return u.f;
}
__device__ __forceinline__ f32x8 wmma_bf16(bf16x16 a, bf16x16 b, f32x8 c) {
  // 8 args: (neg_a, A, neg_b, B, c_mod, C, reuse_a, reuse_b)
  return __builtin_amdgcn_wmma_f32_16x16x32_bf16(false, a, false, b,
                                                 (short)0, c, false, false);
}

// 16-B global -> LDS copy; async path bypasses VGPRs, tracked by ASYNCcnt.
__device__ __forceinline__ void async_copy_b128(const void* gsrc, void* ldst) {
#if ASYNC_LDS
  __builtin_amdgcn_global_load_async_to_lds_b128(
      (__attribute__((address_space(1))) v4i_vs*)gsrc,
      (__attribute__((address_space(3))) v4i_vs*)ldst, 0, 0);
#else
  *(u16x8*)ldst = *(const u16x8*)gsrc;
#endif
}
__device__ __forceinline__ void async_wait0() {
#if ASYNC_LDS
  asm volatile("s_wait_asynccnt 0x0" ::: "memory");
#endif
}

// ---- kernel 1: fp32 -> (hi, lo) bf16 split (x4 vectorized) ----------------

__global__ void split_f32_bf16(const float* __restrict__ x,
                               unsigned short* __restrict__ hi,
                               unsigned short* __restrict__ lo,
                               long n4) {
  long i = (long)blockIdx.x * blockDim.x + threadIdx.x;
  long stride = (long)gridDim.x * blockDim.x;
  for (; i < n4; i += stride) {
    float4 f4 = ((const float4*)x)[i];
    float fa[4] = {f4.x, f4.y, f4.z, f4.w};
    u16x4 h, l;
#pragma unroll
    for (int j = 0; j < 4; j++) {
      unsigned short hb = f32_to_bf16_rne(fa[j]);
      h[j] = hb;
      l[j] = f32_to_bf16_rne(fa[j] - bf16_bits_to_f32(hb));
    }
    ((u16x4*)hi)[i] = h;
    ((u16x4*)lo)[i] = l;
  }
}

// ---- kernel 2: split-precision GEMM  C[M,N] = A[M,K] * B[K,N] -------------
// 128x128x32 tile, 256 threads = 8 waves (4x2 wave grid, 32x64 per wave).
// Double-buffered LDS; A staged async, B transposed through VGPRs in k-pairs.

#define BM 128
#define BN 128
#define BK 32
#define AS_STRIDE 40   // halves; row pitch 80 B (16-B aligned, conflict-free)
#define BS_STRIDE 40

__global__ __launch_bounds__(256)
void gemm_split_bf16(const unsigned short* __restrict__ Ah,
                     const unsigned short* __restrict__ Al,
                     const unsigned short* __restrict__ Bh,
                     const unsigned short* __restrict__ Bl,
                     float* __restrict__ C, int M, int N, int K) {
  __shared__ unsigned short AsH[2][BM * AS_STRIDE], AsL[2][BM * AS_STRIDE];
  __shared__ unsigned short BsH[2][BN * BS_STRIDE], BsL[2][BN * BS_STRIDE];

  const int tid   = threadIdx.x;
  const int lane  = tid & 31;
  const int w     = tid >> 5;
  const int waveM = w >> 1;          // 0..3  -> 32-row strip
  const int waveN = w & 1;           // 0..1  -> 64-col strip
  const int l15   = lane & 15;
  const int lhi   = lane >> 4;
  const int m0    = blockIdx.y * BM;
  const int n0    = blockIdx.x * BN;

  const f32x8 fz = {0.f, 0.f, 0.f, 0.f, 0.f, 0.f, 0.f, 0.f};
  f32x8 acc[2][4];
#pragma unroll
  for (int mt = 0; mt < 2; mt++)
#pragma unroll
    for (int nt = 0; nt < 4; nt++) acc[mt][nt] = fz;

  auto stage = [&](int buf, int k0) {
    // A tile (row-major): pure async global->LDS, 2x b128 per component.
#pragma unroll
    for (int p = 0; p < 2; p++) {
      int idx = tid + p * 256;       // 0..511
      int row = idx >> 2, ch = idx & 3;
      size_t g = (size_t)(m0 + row) * K + k0 + ch * 8;
      async_copy_b128(&Ah[g], &AsH[buf][row * AS_STRIDE + ch * 8]);
      async_copy_b128(&Al[g], &AsL[buf][row * AS_STRIDE + ch * 8]);
    }
    // B tile -> [n][k] transpose via VGPRs, storing packed (k, k+1) pairs.
    {
      int kp = tid >> 4, ch = tid & 15;              // rows 2kp,2kp+1; col chunk
      size_t g0 = (size_t)(k0 + 2 * kp) * N + n0 + ch * 8;
      u16x8 h0 = *(const u16x8*)&Bh[g0];
      u16x8 h1 = *(const u16x8*)&Bh[g0 + N];
      u16x8 l0 = *(const u16x8*)&Bl[g0];
      u16x8 l1 = *(const u16x8*)&Bl[g0 + N];
#pragma unroll
      for (int i = 0; i < 8; i++) {
        int n = ch * 8 + i;
        *(unsigned int*)&BsH[buf][n * BS_STRIDE + 2 * kp] =
            (unsigned int)h0[i] | ((unsigned int)h1[i] << 16);
        *(unsigned int*)&BsL[buf][n * BS_STRIDE + 2 * kp] =
            (unsigned int)l0[i] | ((unsigned int)l1[i] << 16);
      }
    }
  };

  const int KT = K / BK;
  stage(0, 0);
  for (int kt = 0; kt < KT; kt++) {
    const int buf = kt & 1;
    async_wait0();          // our async writes into LDS[buf] have landed
    __syncthreads();        // everyone's writes visible; prev buffer free
    if (kt + 1 < KT) stage(buf ^ 1, (kt + 1) * BK);   // overlap with WMMAs

    // ---- fragments (ISA 16-bit A(16x32) / B(32x16) layouts) ----
    bf16x16 aH[2], aL[2], bH[4], bL[4];
#pragma unroll
    for (int mt = 0; mt < 2; mt++) {
      int r = waveM * 32 + mt * 16 + l15;
      int kh = lhi * 8;                 // lanes 0-15: K0..7,K16..23; 16-31: +8
      aH[mt] = frag16(*(const u16x8*)&AsH[buf][r * AS_STRIDE + kh],
                      *(const u16x8*)&AsH[buf][r * AS_STRIDE + kh + 16]);
      aL[mt] = frag16(*(const u16x8*)&AsL[buf][r * AS_STRIDE + kh],
                      *(const u16x8*)&AsL[buf][r * AS_STRIDE + kh + 16]);
    }
#pragma unroll
    for (int nt = 0; nt < 4; nt++) {
      int n = waveN * 64 + nt * 16 + l15;
      int kh = lhi * 8;                 // lanes 0-15: K0..7,K16..23; 16-31: +8
      bH[nt] = frag16(*(const u16x8*)&BsH[buf][n * BS_STRIDE + kh],
                      *(const u16x8*)&BsH[buf][n * BS_STRIDE + kh + 16]);
      bL[nt] = frag16(*(const u16x8*)&BsL[buf][n * BS_STRIDE + kh],
                      *(const u16x8*)&BsL[buf][n * BS_STRIDE + kh + 16]);
    }
    // ---- 3-term split-precision MACs: Ah*Bh + Ah*Bl + Al*Bh ----
#pragma unroll
    for (int mt = 0; mt < 2; mt++)
#pragma unroll
      for (int nt = 0; nt < 4; nt++) {
        acc[mt][nt] = wmma_bf16(aL[mt], bH[nt], acc[mt][nt]);
        acc[mt][nt] = wmma_bf16(aH[mt], bL[nt], acc[mt][nt]);
        acc[mt][nt] = wmma_bf16(aH[mt], bH[nt], acc[mt][nt]);
      }
  }

  // ---- epilogue: C layout VGPR r -> row (r + 8*(lane>=16)), col lane&15 ----
#pragma unroll
  for (int mt = 0; mt < 2; mt++)
#pragma unroll
    for (int nt = 0; nt < 4; nt++)
#pragma unroll
      for (int r = 0; r < 8; r++) {
        int row = m0 + waveM * 32 + mt * 16 + r + 8 * lhi;
        int col = n0 + waveN * 64 + nt * 16 + l15;
        C[(size_t)row * N + col] = acc[mt][nt][r];
      }
}

// ---- kernel 3: per-head RMSNorm + RoPE + hi/lo split ----------------------
// grid (T, 48): slot 0..31 = q heads, 32..39 = k heads, 40..47 = v heads.

__global__ __launch_bounds__(128)
void norm_rope_split(const float* __restrict__ qkv,
                     const int* __restrict__ positions,
                     const float* __restrict__ qw,
                     const float* __restrict__ kw,
                     unsigned short* __restrict__ qH, unsigned short* __restrict__ qL,
                     unsigned short* __restrict__ kH, unsigned short* __restrict__ kL,
                     unsigned short* __restrict__ vH, unsigned short* __restrict__ vL) {
  const int t = blockIdx.x;
  const int slot = blockIdx.y;
  const int d = threadIdx.x;
  __shared__ float red[4];
  __shared__ float nbuf[HD];

  float val;
  if (slot < 32)       val = qkv[(size_t)t * QKVW + slot * HD + d];
  else if (slot < 40)  val = qkv[(size_t)t * QKVW + 4096 + (slot - 32) * HD + d];
  else                 val = qkv[(size_t)t * QKVW + 5120 + (slot - 40) * HD + d];

  if (slot < 40) {
    // RMSNorm over 128 lanes (4 waves)
    float ss = val * val;
    ss += __shfl_xor(ss, 16, 32);
    ss += __shfl_xor(ss, 8, 32);
    ss += __shfl_xor(ss, 4, 32);
    ss += __shfl_xor(ss, 2, 32);
    ss += __shfl_xor(ss, 1, 32);
    if ((threadIdx.x & 31) == 0) red[threadIdx.x >> 5] = ss;
    __syncthreads();
    float tot = red[0] + red[1] + red[2] + red[3];
    float rms = rsqrtf(tot * (1.0f / (float)HD) + 1e-6f);
    const float* wn = (slot < 32) ? qw : kw;
    float nv = val * rms * wn[d];
    nbuf[d] = nv;
    __syncthreads();
    // RoPE
    int i = d & 63;
    float x1 = nbuf[i], x2 = nbuf[i + 64];
    float inv = __powf(10000.0f, -((float)(2 * i)) * (1.0f / (float)HD));
    float ang = (float)positions[t] * inv;
    float s, c;
    __sincosf(ang, &s, &c);
    float out = (d < 64) ? (x1 * c - x2 * s) : (x2 * c + x1 * s);
    unsigned short hb = f32_to_bf16_rne(out);
    unsigned short lb = f32_to_bf16_rne(out - bf16_bits_to_f32(hb));
    if (slot < 32) {
      size_t o = ((size_t)slot * T_ + t) * HD + d;     // q: [head][t][d]
      qH[o] = hb; qL[o] = lb;
    } else {
      size_t o = ((size_t)(slot - 32) * T_ + t) * HD + d;
      kH[o] = hb; kL[o] = lb;
    }
  } else {
    unsigned short hb = f32_to_bf16_rne(val);
    unsigned short lb = f32_to_bf16_rne(val - bf16_bits_to_f32(hb));
    size_t o = ((size_t)(slot - 40) * T_ + t) * HD + d;
    vH[o] = hb; vL[o] = lb;
  }
}

// ---- kernel 4: causal GQA flash attention ---------------------------------
// grid (32 qblocks, 32 heads); 128 threads = 4 waves, 16 queries per wave;
// 32-key inner blocks, double-buffered K/V tiles (K async, V transposed).

#define KS_STRIDE 136   // 128 + 8 pad halves (272 B row pitch, 16-B aligned)
#define VS_STRIDE 40
#define PS_STRIDE 40

__global__ __launch_bounds__(128)
void flash_attn_split(const unsigned short* __restrict__ qH, const unsigned short* __restrict__ qL,
                      const unsigned short* __restrict__ kH, const unsigned short* __restrict__ kL,
                      const unsigned short* __restrict__ vH, const unsigned short* __restrict__ vL,
                      unsigned short* __restrict__ oH, unsigned short* __restrict__ oL) {
  const int qb = blockIdx.x;
  const int h  = blockIdx.y;
  const int kvh = h >> 2;                 // GQA: 4 q heads per kv head
  const int tid = threadIdx.x;
  const int lane = tid & 31;
  const int w = tid >> 5;
  const int l15 = lane & 15, lhi = lane >> 4;
  const int q0 = qb * 64 + w * 16;        // this wave's 16 query rows

  __shared__ unsigned short KsH[2][32 * KS_STRIDE], KsL[2][32 * KS_STRIDE]; // [key][dim]
  __shared__ unsigned short VsH[2][HD * VS_STRIDE], VsL[2][HD * VS_STRIDE]; // [dim][key]
  __shared__ unsigned short PsH[4 * 16 * PS_STRIDE], PsL[4 * 16 * PS_STRIDE];

  const f32x8 fz = {0.f, 0.f, 0.f, 0.f, 0.f, 0.f, 0.f, 0.f};

  // Q fragments live in registers for the whole kernel (4 k-chunks x hi/lo)
  bf16x16 qfH[4], qfL[4];
  {
    size_t qbase = ((size_t)h * T_ + (q0 + l15)) * HD;
#pragma unroll
    for (int kc = 0; kc < 4; kc++) {
      size_t g = qbase + kc * 32 + lhi * 8;
      qfH[kc] = frag16(*(const u16x8*)&qH[g], *(const u16x8*)&qH[g + 16]);
      qfL[kc] = frag16(*(const u16x8*)&qL[g], *(const u16x8*)&qL[g + 16]);
    }
  }

  f32x8 accO[8];
#pragma unroll
  for (int nt = 0; nt < 8; nt++) accO[nt] = fz;
  float mrow[8], lrow[8];
#pragma unroll
  for (int r = 0; r < 8; r++) { mrow[r] = -1e30f; lrow[r] = 0.f; }

  auto stage = [&](int buf, int kb) {
    // K tile (row-major): async global -> LDS.
#pragma unroll
    for (int p = 0; p < 4; p++) {
      int idx = tid + p * 128;            // 0..511
      int kk = idx >> 4, ch = idx & 15;
      size_t g = ((size_t)kvh * T_ + kb * 32 + kk) * HD + ch * 8;
      async_copy_b128(&kH[g], &KsH[buf][kk * KS_STRIDE + ch * 8]);
      async_copy_b128(&kL[g], &KsL[buf][kk * KS_STRIDE + ch * 8]);
    }
    // V tile -> [dim][key] transpose via VGPRs, packed (k, k+1) pairs.
#pragma unroll
    for (int p = 0; p < 2; p++) {
      int idx = tid + p * 128;            // 0..255
      int kp = idx >> 4, ch = idx & 15;   // keys 2kp,2kp+1; dim chunk
      size_t g0 = ((size_t)kvh * T_ + kb * 32 + 2 * kp) * HD + ch * 8;
      u16x8 h0 = *(const u16x8*)&vH[g0];
      u16x8 h1 = *(const u16x8*)&vH[g0 + HD];
      u16x8 l0 = *(const u16x8*)&vL[g0];
      u16x8 l1 = *(const u16x8*)&vL[g0 + HD];
#pragma unroll
      for (int i = 0; i < 8; i++) {
        int n = ch * 8 + i;
        *(unsigned int*)&VsH[buf][n * VS_STRIDE + 2 * kp] =
            (unsigned int)h0[i] | ((unsigned int)h1[i] << 16);
        *(unsigned int*)&VsL[buf][n * VS_STRIDE + 2 * kp] =
            (unsigned int)l0[i] | ((unsigned int)l1[i] << 16);
      }
    }
  };

  const int nkb = 2 * qb + 2;             // causal: keys <= qb*64+63
  stage(0, 0);
  for (int kb = 0; kb < nkb; kb++) {
    const int buf = kb & 1;
    async_wait0();
    __syncthreads();
    if (kb + 1 < nkb) stage(buf ^ 1, kb + 1);   // overlap with S/softmax/PV

    // ---- S = Q * K^T  (16 queries x 32 keys), 3-term split ----
    f32x8 s[2];
    s[0] = fz; s[1] = fz;
#pragma unroll
    for (int nt = 0; nt < 2; nt++) {
      int key = nt * 16 + l15;
#pragma unroll
      for (int kc = 0; kc < 4; kc++) {
        int kh = kc * 32 + lhi * 8;
        bf16x16 bHf = frag16(*(const u16x8*)&KsH[buf][key * KS_STRIDE + kh],
                             *(const u16x8*)&KsH[buf][key * KS_STRIDE + kh + 16]);
        bf16x16 bLf = frag16(*(const u16x8*)&KsL[buf][key * KS_STRIDE + kh],
                             *(const u16x8*)&KsL[buf][key * KS_STRIDE + kh + 16]);
        s[nt] = wmma_bf16(qfL[kc], bHf, s[nt]);
        s[nt] = wmma_bf16(qfH[kc], bLf, s[nt]);
        s[nt] = wmma_bf16(qfH[kc], bHf, s[nt]);
      }
    }

    // ---- online softmax (rows live in 16-lane halves of the wave) ----
    float alpha[8];
#pragma unroll
    for (int r = 0; r < 8; r++) {
      int qrow = q0 + r + 8 * lhi;
      float s0 = s[0][r] * SCALE;
      float s1 = s[1][r] * SCALE;
      int key0 = kb * 32 + l15;
      if (key0 > qrow)       s0 = -1e30f;
      if (key0 + 16 > qrow)  s1 = -1e30f;
      float mx = fmaxf(s0, s1);
      mx = fmaxf(mx, __shfl_xor(mx, 8, 16));
      mx = fmaxf(mx, __shfl_xor(mx, 4, 16));
      mx = fmaxf(mx, __shfl_xor(mx, 2, 16));
      mx = fmaxf(mx, __shfl_xor(mx, 1, 16));
      float mn = fmaxf(mrow[r], mx);
      float a  = __expf(mrow[r] - mn);
      float p0 = __expf(s0 - mn);
      float p1 = __expf(s1 - mn);
      float ps = p0 + p1;
      ps += __shfl_xor(ps, 8, 16);
      ps += __shfl_xor(ps, 4, 16);
      ps += __shfl_xor(ps, 2, 16);
      ps += __shfl_xor(ps, 1, 16);
      mrow[r] = mn;
      lrow[r] = lrow[r] * a + ps;
      alpha[r] = a;
      // stash P (hi/lo) into per-wave LDS for A-fragment re-layout
      int m = r + 8 * lhi;
      unsigned short h0 = f32_to_bf16_rne(p0);
      unsigned short h1 = f32_to_bf16_rne(p1);
      PsH[(w * 16 + m) * PS_STRIDE + l15]      = h0;
      PsH[(w * 16 + m) * PS_STRIDE + 16 + l15] = h1;
      PsL[(w * 16 + m) * PS_STRIDE + l15]      = f32_to_bf16_rne(p0 - bf16_bits_to_f32(h0));
      PsL[(w * 16 + m) * PS_STRIDE + 16 + l15] = f32_to_bf16_rne(p1 - bf16_bits_to_f32(h1));
    }
    __syncthreads();

    // ---- P fragments (A layout) ----
    bf16x16 pH, pL;
    {
      int m = w * 16 + l15;
      int kh = lhi * 8;
      pH = frag16(*(const u16x8*)&PsH[m * PS_STRIDE + kh],
                  *(const u16x8*)&PsH[m * PS_STRIDE + kh + 16]);
      pL = frag16(*(const u16x8*)&PsL[m * PS_STRIDE + kh],
                  *(const u16x8*)&PsL[m * PS_STRIDE + kh + 16]);
    }

    // ---- O = O*alpha + P*V  (contraction over 32 keys) ----
#pragma unroll
    for (int nt = 0; nt < 8; nt++) {
#pragma unroll
      for (int r = 0; r < 8; r++) accO[nt][r] *= alpha[r];
      int n = nt * 16 + l15;
      int kh = lhi * 8;
      bf16x16 vfH = frag16(*(const u16x8*)&VsH[buf][n * VS_STRIDE + kh],
                           *(const u16x8*)&VsH[buf][n * VS_STRIDE + kh + 16]);
      bf16x16 vfL = frag16(*(const u16x8*)&VsL[buf][n * VS_STRIDE + kh],
                           *(const u16x8*)&VsL[buf][n * VS_STRIDE + kh + 16]);
      accO[nt] = wmma_bf16(pL, vfH, accO[nt]);
      accO[nt] = wmma_bf16(pH, vfL, accO[nt]);
      accO[nt] = wmma_bf16(pH, vfH, accO[nt]);
    }
  }

  // ---- epilogue: normalize, re-split to bf16 hi/lo for the O-projection ----
#pragma unroll
  for (int nt = 0; nt < 8; nt++)
#pragma unroll
    for (int r = 0; r < 8; r++) {
      int qrow = q0 + r + 8 * lhi;
      int col = h * HD + nt * 16 + l15;
      float ov = accO[nt][r] / lrow[r];
      size_t o = (size_t)qrow * HID + col;
      unsigned short hb = f32_to_bf16_rne(ov);
      oH[o] = hb;
      oL[o] = f32_to_bf16_rne(ov - bf16_bits_to_f32(hb));
    }
}

// ---------------------------------------------------------------------------
// host launcher
// ---------------------------------------------------------------------------

extern "C" void kernel_launch(void* const* d_in, const int* in_sizes, int n_in,
                              void* d_out, int out_size, void* d_ws, size_t ws_size,
                              hipStream_t stream) {
  const int*   positions = (const int*)  d_in[0];
  const float* hidden    = (const float*)d_in[1];
  const float* w_qkv     = (const float*)d_in[2];
  const float* w_o       = (const float*)d_in[3];
  const float* q_norm_w  = (const float*)d_in[4];
  const float* k_norm_w  = (const float*)d_in[5];

  const size_t HS_N   = (size_t)T_ * HID;        // 8,388,608
  const size_t WQKV_N = (size_t)HID * QKVW;      // 25,165,824
  const size_t WO_N   = (size_t)HID * HID;       // 16,777,216
  const size_t QKV_N  = (size_t)T_ * QKVW;       // 12,582,912
  const size_t KV_N   = (size_t)NKVH * T_ * HD;  // 2,097,152

  char* p = (char*)d_ws;
  auto alloc = [&](size_t bytes) -> char* {
    char* r = p;
    p += (bytes + 255) & ~(size_t)255;
    return r;
  };
  unsigned short* hsH   = (unsigned short*)alloc(HS_N * 2);
  unsigned short* hsL   = (unsigned short*)alloc(HS_N * 2);
  unsigned short* wqkvH = (unsigned short*)alloc(WQKV_N * 2);
  unsigned short* wqkvL = (unsigned short*)alloc(WQKV_N * 2);
  unsigned short* woH   = (unsigned short*)alloc(WO_N * 2);
  unsigned short* woL   = (unsigned short*)alloc(WO_N * 2);
  float*          qkv   = (float*)alloc(QKV_N * 4);
  unsigned short* kHp   = (unsigned short*)alloc(KV_N * 2);
  unsigned short* kLp   = (unsigned short*)alloc(KV_N * 2);
  unsigned short* vHp   = (unsigned short*)alloc(KV_N * 2);
  unsigned short* vLp   = (unsigned short*)alloc(KV_N * 2);
  if ((size_t)(p - (char*)d_ws) > ws_size) return;   // ws too small: bail
  // stream-ordered aliases (dead buffers reused):
  unsigned short* qHp = hsH;    // q [32][2048][128]  == size of hidden split
  unsigned short* qLp = hsL;
  unsigned short* oHp = wqkvH;  // attention out [2048][4096] fits in wqkv split
  unsigned short* oLp = wqkvL;

  // 1) precision-split inputs (n divisible by 4 for all three)
  split_f32_bf16<<<4096, 256, 0, stream>>>(hidden, hsH,   hsL,   (long)(HS_N / 4));
  split_f32_bf16<<<4096, 256, 0, stream>>>(w_qkv,  wqkvH, wqkvL, (long)(WQKV_N / 4));
  split_f32_bf16<<<4096, 256, 0, stream>>>(w_o,    woH,   woL,   (long)(WO_N / 4));

  // 2) QKV projection: [2048,4096] x [4096,6144] -> fp32 qkv
  gemm_split_bf16<<<dim3(QKVW / BN, T_ / BM), 256, 0, stream>>>(
      hsH, hsL, wqkvH, wqkvL, qkv, T_, QKVW, HID);

  // 3) RMSNorm + RoPE + re-split (q uses hidden-split buffers, now dead)
  norm_rope_split<<<dim3(T_, NQH + 2 * NKVH), 128, 0, stream>>>(
      qkv, positions, q_norm_w, k_norm_w, qHp, qLp, kHp, kLp, vHp, vLp);

  // 4) causal GQA flash attention -> bf16 hi/lo O (wqkv split buffers, now dead)
  flash_attn_split<<<dim3(T_ / 64, NQH), 128, 0, stream>>>(
      qHp, qLp, kHp, kLp, vHp, vLp, oHp, oLp);

  // 5) output projection: [2048,4096] x [4096,4096] -> fp32 d_out
  gemm_split_bf16<<<dim3(HID / BN, T_ / BM), 256, 0, stream>>>(
      oHp, oLp, woH, woL, (float*)d_out, T_, HID, HID);
}
